// MultiheadSelfAttention_20272245637212
// MI455X (gfx1250) — matrix operand order repaired and checked by reading, not compiled, not run on hardware
//
#include <hip/hip_runtime.h>

// ---------------- problem constants ----------------
#define D_MODEL   1024
#define NUM_HEADS 16
#define D_K       64
#define SEQ       2048
#define BATCH     2
#define MROWS     (BATCH * SEQ)   // 4096

// ---------------- vector types for WMMA ----------------
typedef __attribute__((ext_vector_type(4)))  __bf16 bf16x4;
typedef __attribute__((ext_vector_type(8)))  __bf16 bf16x8;
typedef __attribute__((ext_vector_type(16))) __bf16 bf16x16;
typedef __attribute__((ext_vector_type(8)))  float  floatx8;

// ---------------- async global->LDS copy support (CDNA5) ----------------
#if defined(__has_builtin)
#if __has_builtin(__builtin_amdgcn_global_load_async_to_lds_b128)
#define HAVE_ASYNC_LDS 1
#endif
#endif
#ifndef HAVE_ASYNC_LDS
#define HAVE_ASYNC_LDS 0
#endif

#if HAVE_ASYNC_LDS
typedef int v4i __attribute__((vector_size(16)));
typedef __attribute__((address_space(1))) v4i* glob_v4i_ptr;   // global-AS v4i*
typedef __attribute__((address_space(3))) v4i* lds_v4i_ptr;    // LDS-AS v4i*
#endif

static __device__ __forceinline__ __bf16 f2bf(float f) {
    unsigned u = __builtin_bit_cast(unsigned, f);
    u += 0x7fffu + ((u >> 16) & 1u);                 // round-to-nearest-even
    unsigned short h = (unsigned short)(u >> 16);
    return __builtin_bit_cast(__bf16, h);
}
static __device__ __forceinline__ float bf2f(__bf16 b) {
    unsigned u = ((unsigned)__builtin_bit_cast(unsigned short, b)) << 16;
    return __builtin_bit_cast(float, u);
}
static __device__ __forceinline__ bf16x8 load8(const __bf16* p) {
    return *(const bf16x8*)p;                        // global_load_b128 / ds_load_b128
}
static __device__ __forceinline__ void store8(__bf16* p, bf16x8 v) {
    *(bf16x8*)p = v;
}
static __device__ __forceinline__ bf16x16 mk16(bf16x8 lo, bf16x8 hi) {
    bf16x16 r;
#pragma unroll
    for (int i = 0; i < 8; ++i) { r[i] = lo[i]; r[i + 8] = hi[i]; }
    return r;
}
static __device__ __forceinline__ floatx8 wmma_bf16(bf16x16 a, bf16x16 b, floatx8 c) {
    // D = A(16x32) * B(32x16) + C, f32 accumulate
    return __builtin_amdgcn_wmma_f32_16x16x32_bf16(false, a, false, b, (short)0, c,
                                                   false, false);
}

// 16-byte global -> LDS copy: async (TA writes LDS directly, ASYNCcnt) or
// fallback through VGPRs.
static __device__ __forceinline__ void copy128(const __bf16* g, __bf16* l) {
#if HAVE_ASYNC_LDS
    __builtin_amdgcn_global_load_async_to_lds_b128((glob_v4i_ptr)g, (lds_v4i_ptr)l,
                                                   0, 0);
#else
    store8(l, load8(g));
#endif
}
static __device__ __forceinline__ void async_wait() {
#if HAVE_ASYNC_LDS
    asm volatile("s_wait_asynccnt 0x0" ::: "memory");
#endif
}

// ---------------- f32 -> bf16 conversion (vectorized) ----------------
__global__ void cvt_f32_bf16(const float* __restrict__ in, __bf16* __restrict__ out,
                             int n4) {
    int i = blockIdx.x * blockDim.x + threadIdx.x;
    if (i >= n4) return;
    const float4 v = ((const float4*)in)[i];
    bf16x4 o;
    o[0] = f2bf(v.x); o[1] = f2bf(v.y); o[2] = f2bf(v.z); o[3] = f2bf(v.w);
    ((bf16x4*)out)[i] = o;
}

// ---------------- GEMM: Y[M,N] = A[M,1024] * W[N,1024]^T --------------
// MODE 0: write bf16 scatter [B,H,S,dk]         (Q, K)
// MODE 1: write bf16 scatter [B,H,dk,S] (V transposed for flash B-operand)
// MODE 2: write f32 row-major [M,N] to d_out    (output projection)
// Block = 256 threads = 8 waves arranged 4(M) x 2(N); WG tile 128x128.
template <int MODE>
__global__ __launch_bounds__(256) void gemm_wmma(const __bf16* __restrict__ A,
                                                 const __bf16* __restrict__ W,
                                                 void* __restrict__ Yv) {
    const int lane = threadIdx.x & 31;
    const int wave = threadIdx.x >> 5;
    const int ln   = lane & 15;
    const int lh   = lane >> 4;          // lane half (0/1)
    const int m0   = blockIdx.x * 128 + (wave >> 1) * 32;
    const int n0   = blockIdx.y * 128 + (wave & 1) * 64;
    const int K    = D_MODEL;
    const int koffA = lh * 8;            // A-frag: halves K kb..kb+7, 16+kb..23+kb
    const int kbB   = lh * 8;            // B-frag: halves K kb..kb+7, 16+kb..23+kb

    floatx8 acc[2][4];
#pragma unroll
    for (int im = 0; im < 2; ++im)
#pragma unroll
        for (int in = 0; in < 4; ++in)
#pragma unroll
            for (int e = 0; e < 8; ++e) acc[im][in][e] = 0.f;

    for (int kk = 0; kk < K; kk += 32) {
        bf16x16 af[2];
#pragma unroll
        for (int im = 0; im < 2; ++im) {
            const __bf16* p = A + (size_t)(m0 + im * 16 + ln) * K + kk;
            __builtin_prefetch(p + 32, 0, 0);    // next K-tile of A
            af[im] = mk16(load8(p + koffA), load8(p + 16 + koffA));
        }
#pragma unroll
        for (int in = 0; in < 4; ++in) {
            const __bf16* p = W + (size_t)(n0 + in * 16 + ln) * K + kk + kbB;
            bf16x16 bfr = mk16(load8(p), load8(p + 16));
#pragma unroll
            for (int im = 0; im < 2; ++im)
                acc[im][in] = wmma_bf16(af[im], bfr, acc[im][in]);
        }
    }

    // Epilogue: C layout — VGPR e holds row (e + 8*laneHalf), col = lane&15
#pragma unroll
    for (int im = 0; im < 2; ++im) {
#pragma unroll
        for (int in = 0; in < 4; ++in) {
#pragma unroll
            for (int e = 0; e < 8; ++e) {
                const int row = m0 + im * 16 + e + lh * 8;  // b*SEQ + s
                const int col = n0 + in * 16 + ln;          // h*D_K + dk
                const float v = acc[im][in][e];
                if (MODE == 2) {
                    ((float*)Yv)[(size_t)row * D_MODEL + col] = v;
                } else {
                    const int b = row >> 11, s = row & (SEQ - 1);
                    const int h = col >> 6, d = col & (D_K - 1);
                    __bf16* Y = (__bf16*)Yv;
                    if (MODE == 0)
                        Y[((size_t)(b * NUM_HEADS + h) * SEQ + s) * D_K + d] = f2bf(v);
                    else
                        Y[((size_t)(b * NUM_HEADS + h) * D_K + d) * SEQ + s] = f2bf(v);
                }
            }
        }
    }
}

// ---------------- RoPE (interleaved pairs) on Q and K in-place ----------------
__global__ void rope_kernel(__bf16* __restrict__ q, __bf16* __restrict__ k,
                            const int* __restrict__ pos) {
    const int total = BATCH * NUM_HEADS * SEQ * (D_K / 2);  // 2,097,152
    int idx = blockIdx.x * blockDim.x + threadIdx.x;
    if (idx >= total) return;
    const int p = idx & 31;            // pair index 0..31
    const int s = (idx >> 5) & (SEQ - 1);
    const int h = (idx >> 16) & (NUM_HEADS - 1);
    const int b = idx >> 20;
    const int pv = pos[b * SEQ + s];
    const float inv = __powf(10000.f, -(float)(2 * p) / (float)D_K);
    float sn, cs;
    __sincosf((float)pv * inv, &sn, &cs);
    const size_t base = (((size_t)(b * NUM_HEADS + h)) * SEQ + s) * D_K + 2 * p;

    float xe = bf2f(q[base]), xo = bf2f(q[base + 1]);
    q[base]     = f2bf(xe * cs - xo * sn);
    q[base + 1] = f2bf(xe * sn + xo * cs);
    xe = bf2f(k[base]); xo = bf2f(k[base + 1]);
    k[base]     = f2bf(xe * cs - xo * sn);
    k[base + 1] = f2bf(xe * sn + xo * cs);
}

// ---------------- online-softmax step (templated on masking) ----------------
template <bool MASKED>
static __device__ __forceinline__ void online_step(floatx8& s0, floatx8& s1,
                                                   floatx8 (&o)[4],
                                                   float (&mrow)[8], float (&lrow)[8],
                                                   __bf16* myP, int qbase, int j0,
                                                   int ln, int lh) {
    const float sc = 0.125f;                         // 1/sqrt(64)
#pragma unroll
    for (int e = 0; e < 8; ++e) {
        const int qpos = qbase + e + lh * 8;
        float v0 = s0[e] * sc;
        float v1 = s1[e] * sc;
        if (MASKED) {
            if (j0 + ln      > qpos) v0 = -1e30f;
            if (j0 + 16 + ln > qpos) v1 = -1e30f;
        }
        float mx = fmaxf(v0, v1);
        mx = fmaxf(mx, __shfl_xor(mx, 1));
        mx = fmaxf(mx, __shfl_xor(mx, 2));
        mx = fmaxf(mx, __shfl_xor(mx, 4));
        mx = fmaxf(mx, __shfl_xor(mx, 8));           // masks keep each 16-lane half
        const float mn   = fmaxf(mrow[e], mx);
        const float corr = __expf(mrow[e] - mn);
        const float p0 = __expf(v0 - mn);
        const float p1 = __expf(v1 - mn);
        float rs = p0 + p1;
        rs += __shfl_xor(rs, 1);
        rs += __shfl_xor(rs, 2);
        rs += __shfl_xor(rs, 4);
        rs += __shfl_xor(rs, 8);
        lrow[e] = lrow[e] * corr + rs;
        mrow[e] = mn;
#pragma unroll
        for (int d = 0; d < 4; ++d) o[d][e] *= corr;
        const int r = e + lh * 8;                    // C-layout row
        myP[r * 32 + ln]      = f2bf(p0);
        myP[r * 32 + 16 + ln] = f2bf(p1);
    }
}

// ---------------- causal flash attention ----------------
// grid = (SEQ/128, BATCH*NUM_HEADS), block = 256 (8 waves, 16 q-rows each).
// Q,K in [B,H,S,dk] bf16; V transposed [B,H,dk,S] bf16; Out bf16 [B*S, D_MODEL].
// K/V tiles double-buffered in LDS, filled by async global->LDS copies issued
// one block ahead (ASYNCcnt) so the copy overlaps the 8 WMMAs + softmax.
__global__ __launch_bounds__(256) void flash_kernel(const __bf16* __restrict__ Q,
                                                    const __bf16* __restrict__ Km,
                                                    const __bf16* __restrict__ Vt,
                                                    __bf16* __restrict__ Out) {
    __shared__ __bf16 ldsK[2][32 * D_K];         // [buf][key][feature]  2 x 4 KB
    __shared__ __bf16 ldsV[2][D_K * 32];         // [buf][dim][key]      2 x 4 KB
    __shared__ __bf16 ldsP[8][16 * 32];          // per-wave P tile      8 KB

    const int tid   = threadIdx.x;
    const int lane  = tid & 31;
    const int wave  = tid >> 5;
    const int ln    = lane & 15;
    const int lh    = lane >> 4;
    const int bh    = blockIdx.y;
    const int qbase = blockIdx.x * 128 + wave * 16;

    const __bf16* Qp = Q  + (size_t)bh * SEQ * D_K;
    const __bf16* Kp = Km + (size_t)bh * SEQ * D_K;
    const __bf16* Vp = Vt + (size_t)bh * D_K * SEQ;

    // Q A-fragments (16 rows x 64 features = 2 fragments), loaded once
    bf16x16 qa[2];
    {
        const __bf16* p = Qp + (size_t)(qbase + ln) * D_K;
        const int koff = lh * 8;
        qa[0] = mk16(load8(p + koff),      load8(p + 16 + koff));
        qa[1] = mk16(load8(p + 32 + koff), load8(p + 48 + koff));
    }

    floatx8 o[4];
    float mrow[8], lrow[8];
#pragma unroll
    for (int d = 0; d < 4; ++d)
#pragma unroll
        for (int e = 0; e < 8; ++e) o[d][e] = 0.f;
#pragma unroll
    for (int e = 0; e < 8; ++e) { mrow[e] = -1e30f; lrow[e] = 0.f; }

    // WG-uniform block count (max over the 8 waves' causal bounds)
    const int nblocksWG = (blockIdx.x * 128 + 112 + 16 + 31) >> 5;
    // blocks entirely below the diagonal for THIS wave (no masking needed)
    const int nfull = (qbase + 1) >> 5;
    const int kbB = lh * 8;
    __bf16* myP = &ldsP[wave][0];

    // staging assignments (one 16B copy into each of K/V tile per thread)
    const int kRow = tid >> 3, kSeg = tid & 7;   // K: 32 rows x 8 segs
    const int vDim = tid >> 2, vSeg = tid & 3;   // V: 64 dims x 4 segs
    const __bf16* gK = Kp + (size_t)kRow * D_K + kSeg * 8;   // + j0*D_K
    const __bf16* gV = Vp + (size_t)vDim * SEQ + vSeg * 8;   // + j0

    // prologue: stage tile 0 into buffer 0
    copy128(gK, &ldsK[0][kRow * D_K + kSeg * 8]);
    copy128(gV, &ldsV[0][vDim * 32 + vSeg * 8]);
    async_wait();
    __syncthreads();

    for (int jb = 0; jb < nblocksWG; ++jb) {
        const int j0  = jb * 32;
        const int cur = jb & 1;

        // ---- issue async copy of the NEXT tile into the other buffer ----
        if (jb + 1 < nblocksWG) {
            const int jn = j0 + 32;
            copy128(gK + (size_t)jn * D_K, &ldsK[cur ^ 1][kRow * D_K + kSeg * 8]);
            copy128(gV + jn,               &ldsV[cur ^ 1][vDim * 32 + vSeg * 8]);
        }
        const __bf16* tK = ldsK[cur];
        const __bf16* tV = ldsV[cur];

        // ---- scores: S0 = keys j0..j0+15, S1 = keys j0+16..j0+31 ----
        floatx8 s0, s1;
#pragma unroll
        for (int e = 0; e < 8; ++e) { s0[e] = 0.f; s1[e] = 0.f; }
        {
            const __bf16* kp0 = tK + ln * D_K;
            const __bf16* kp1 = tK + (16 + ln) * D_K;
            bf16x16 b00 = mk16(load8(kp0 + kbB),      load8(kp0 + kbB + 16));
            bf16x16 b01 = mk16(load8(kp0 + 32 + kbB), load8(kp0 + 32 + kbB + 16));
            s0 = wmma_bf16(qa[0], b00, s0);
            s0 = wmma_bf16(qa[1], b01, s0);
            bf16x16 b10 = mk16(load8(kp1 + kbB),      load8(kp1 + kbB + 16));
            bf16x16 b11 = mk16(load8(kp1 + 32 + kbB), load8(kp1 + 32 + kbB + 16));
            s1 = wmma_bf16(qa[0], b10, s1);
            s1 = wmma_bf16(qa[1], b11, s1);
        }

        // ---- online softmax: wave-uniform branch keeps masking off hot path ----
        if (jb < nfull)
            online_step<false>(s0, s1, o, mrow, lrow, myP, qbase, j0, ln, lh);
        else
            online_step<true>(s0, s1, o, mrow, lrow, myP, qbase, j0, ln, lh);

        // wave-local LDS visibility for the P tile (per-wave buffer)
        asm volatile("s_wait_dscnt 0" ::: "memory");
        __builtin_amdgcn_wave_barrier();

        // P as A-fragment (16x32), two b128 LDS loads
        bf16x16 pa;
        {
            const __bf16* p = myP + ln * 32 + lh * 8;
            pa = mk16(load8(p), load8(p + 16));
        }

        // ---- O += P * V (V tile in LDS, contiguous over key dim) ----
#pragma unroll
        for (int d = 0; d < 4; ++d) {
            const __bf16* vp = tV + (d * 16 + ln) * 32 + kbB;
            bf16x16 vb = mk16(load8(vp), load8(vp + 16));
            o[d] = wmma_bf16(pa, vb, o[d]);
        }

        // ---- publish next tile: my async copies done, then WG barrier ----
        async_wait();
        __syncthreads();
    }

    // ---- normalize and write bf16 [B*S, D_MODEL] for the output projection ----
    const int b = bh >> 4, h = bh & (NUM_HEADS - 1);
#pragma unroll
    for (int d = 0; d < 4; ++d) {
#pragma unroll
        for (int e = 0; e < 8; ++e) {
            const int srow = qbase + e + lh * 8;
            const float v  = o[d][e] / lrow[e];
            Out[((size_t)(b * SEQ + srow)) * D_MODEL + h * D_K + d * 16 + ln] = f2bf(v);
        }
    }
}

// ---------------- host-side launch ----------------
extern "C" void kernel_launch(void* const* d_in, const int* in_sizes, int n_in,
                              void* d_out, int out_size, void* d_ws, size_t ws_size,
                              hipStream_t stream) {
    (void)in_sizes; (void)n_in; (void)out_size; (void)ws_size;

    const float* x  = (const float*)d_in[0];
    const float* wq = (const float*)d_in[1];
    const float* wk = (const float*)d_in[2];
    const float* wv = (const float*)d_in[3];
    const float* wo = (const float*)d_in[4];
    const int* pos  = (const int*)d_in[5];
    float* out = (float*)d_out;

    char* ws = (char*)d_ws;
    size_t off = 0;
    const size_t xN  = (size_t)MROWS * D_MODEL;                 // 4M elems
    const size_t wN  = (size_t)D_MODEL * D_MODEL;               // 1M elems
    const size_t hN  = (size_t)BATCH * NUM_HEADS * SEQ * D_K;   // 4M elems
    __bf16* xb   = (__bf16*)(ws + off); off += xN * 2;
    __bf16* wqb  = (__bf16*)(ws + off); off += wN * 2;
    __bf16* wkb  = (__bf16*)(ws + off); off += wN * 2;
    __bf16* wvb  = (__bf16*)(ws + off); off += wN * 2;
    __bf16* wob  = (__bf16*)(ws + off); off += wN * 2;
    __bf16* Qb   = (__bf16*)(ws + off); off += hN * 2;
    __bf16* Kb   = (__bf16*)(ws + off); off += hN * 2;
    __bf16* Vtb  = (__bf16*)(ws + off); off += hN * 2;
    __bf16* attn = (__bf16*)(ws + off); off += xN * 2;          // 48 MB total

    // 1) convert to bf16
    {
        const int n4x = (int)(xN / 4), n4w = (int)(wN / 4);
        cvt_f32_bf16<<<(n4x + 255) / 256, 256, 0, stream>>>(x,  xb,  n4x);
        cvt_f32_bf16<<<(n4w + 255) / 256, 256, 0, stream>>>(wq, wqb, n4w);
        cvt_f32_bf16<<<(n4w + 255) / 256, 256, 0, stream>>>(wk, wkb, n4w);
        cvt_f32_bf16<<<(n4w + 255) / 256, 256, 0, stream>>>(wv, wvb, n4w);
        cvt_f32_bf16<<<(n4w + 255) / 256, 256, 0, stream>>>(wo, wob, n4w);
    }

    // 2) Q/K/V projections (WMMA)
    dim3 ggrid(MROWS / 128, D_MODEL / 128);
    gemm_wmma<0><<<ggrid, 256, 0, stream>>>(xb, wqb, (void*)Qb);
    gemm_wmma<0><<<ggrid, 256, 0, stream>>>(xb, wkb, (void*)Kb);
    gemm_wmma<1><<<ggrid, 256, 0, stream>>>(xb, wvb, (void*)Vtb);   // V transposed

    // 3) RoPE on Q, K
    {
        const int total = BATCH * NUM_HEADS * SEQ * (D_K / 2);
        rope_kernel<<<(total + 255) / 256, 256, 0, stream>>>(Qb, Kb, pos);
    }

    // 4) fused causal flash attention (WMMA, async double-buffered K/V)
    dim3 fgrid(SEQ / 128, BATCH * NUM_HEADS);
    flash_kernel<<<fgrid, 256, 0, stream>>>(Qb, Kb, Vtb, attn);

    // 5) output projection -> f32 d_out
    gemm_wmma<2><<<ggrid, 256, 0, stream>>>(attn, wob, (void*)out);
}
